// PagedAttentionOp_53412213293063
// MI455X (gfx1250) — matrix operand order repaired and checked by reading, not compiled, not run on hardware
//
#include <hip/hip_runtime.h>
#include <hip/hip_bf16.h>

typedef __attribute__((ext_vector_type(16))) __bf16 v16bf;
typedef __attribute__((ext_vector_type(8)))  __bf16 v8bf;
typedef __attribute__((ext_vector_type(8)))  float  v8f;
typedef __attribute__((ext_vector_type(4)))  float  f32x4;

#define NUM_B    32
#define NUM_H    32
#define DIM      128
#define NKV      8
#define NPAGES   8192
#define TPP      16
#define PSLOT    256
#define GRP      4      // query heads per kv head
#define NWAVES   8
#define CHUNK    32     // tokens per iteration (2 pages)
#define NEG_BIG  (-1e10f)

__launch_bounds__(256, 1)
__global__ void paged_attn_wmma(const float* __restrict__ q,      // [B,1,H,D]
                                const float* __restrict__ kp,     // [NKV,NPAGES,TPP,D]
                                const float* __restrict__ vp,     // [NKV,NPAGES,TPP,D]
                                const int*   __restrict__ kvlens, // [B]
                                const int*   __restrict__ pidx,   // [B,PSLOT]
                                float* __restrict__ out)          // [B,1,H,D]
{
  // per-wave V staging tile filled by global_load_async_to_lds_b128
  __shared__ float  v_stage[NWAVES][CHUNK][DIM];      // 128 KB
  __shared__ __bf16 p_lds[NWAVES][16][CHUNK];         // P tile: [row][token], 8 KB
  __shared__ float  o_lds[NWAVES][GRP][DIM];          // 16 KB
  __shared__ float  ml_lds[NWAVES][2][GRP];           // 256 B

  const int b    = blockIdx.x / NKV;
  const int kv   = blockIdx.x % NKV;
  const int wave = threadIdx.x >> 5;
  const int lane = threadIdx.x & 31;
  const int half = lane >> 4;     // wave32 half: K-range of B frags / M-range of C
  const int ln   = lane & 15;

  const int kvlen   = kvlens[b];
  const int nchunks = (kvlen + CHUNK - 1) / CHUNK;

  // ---------- Q -> A fragments (16x32 bf16, rows 0..3 real heads, 4..15 zero) ----------
  // A layout: lane ln row M=ln; elem e<8 -> K = half*8 + e ; e>=8 -> K = 16 + half*8 + (e-8)
  v16bf aq[4];
  {
    const float* qrow = q + ((size_t)b * NUM_H + kv * GRP + (ln < GRP ? ln : 0)) * DIM;
    #pragma unroll
    for (int dc = 0; dc < 4; ++dc) {
      const int d0 = dc * 32 + half * 8;
      f32x4 r0 = *(const f32x4*)(qrow + d0);
      f32x4 r1 = *(const f32x4*)(qrow + d0 + 4);
      f32x4 r2 = *(const f32x4*)(qrow + d0 + 16);
      f32x4 r3 = *(const f32x4*)(qrow + d0 + 20);
      v16bf a;
      #pragma unroll
      for (int e = 0; e < 4; ++e) {
        a[e]      = (__bf16)r0[e];
        a[e + 4]  = (__bf16)r1[e];
        a[e + 8]  = (__bf16)r2[e];
        a[e + 12] = (__bf16)r3[e];
      }
      if (ln >= GRP) {
        #pragma unroll
        for (int e = 0; e < 16; ++e) a[e] = (__bf16)0.0f;
      }
      aq[dc] = a;
    }
  }

  const v8f vzero = {0.f, 0.f, 0.f, 0.f, 0.f, 0.f, 0.f, 0.f};
  float m_run[8], l_run[8];
  v8f   o_acc[8];
  #pragma unroll
  for (int r = 0; r < 8; ++r) { m_run[r] = -INFINITY; l_run[r] = 0.0f; }
  #pragma unroll
  for (int nc = 0; nc < 8; ++nc) o_acc[nc] = vzero;

  const int* my_pages = pidx + (size_t)b * PSLOT;
  const uint32_t vstage_base = (uint32_t)(uintptr_t)(&v_stage[wave][0][0]);

  for (int c = wave; c < nchunks; c += NWAVES) {
    const int cbase = c * CHUNK;
    const int pg0 = my_pages[2 * c];
    const int pg1 = my_pages[2 * c + 1];

    // prefetch this wave's next chunk (global_prefetch_b8)
    if (c + NWAVES < nchunks) {
      const int pn = my_pages[2 * (c + NWAVES)];
      __builtin_prefetch(kp + (((size_t)kv * NPAGES + pn) * TPP) * DIM, 0, 1);
      __builtin_prefetch(vp + (((size_t)kv * NPAGES + pn) * TPP) * DIM, 0, 1);
    }

    // ---------- async-stage V tile (32 tok x 128 f32) into per-wave LDS ----------
    // drain our own ds reads of the previous iteration before overwriting the buffer
    asm volatile("s_wait_dscnt 0x0" ::: "memory");
    {
      const float* vb0 = vp + (((size_t)kv * NPAGES + pg0) * TPP) * DIM;
      const float* vb1 = vp + (((size_t)kv * NPAGES + pg1) * TPP) * DIM;
      #pragma unroll
      for (int i = 0; i < CHUNK; ++i) {   // instruction i stages token i; lane covers d=lane*4..+3
        const float* gsrc = (i < TPP ? vb0 + i * DIM : vb1 + (i - TPP) * DIM) + lane * 4;
        const uint32_t ldsa = vstage_base + (uint32_t)(i * (DIM * 4) + lane * 16);
        asm volatile("global_load_async_to_lds_b128 %0, %1, off"
                     :: "v"(ldsa), "v"(gsrc) : "memory");
      }
    }

    // ---------- scores: S = Q * K^T for two 16-token groups (overlaps async copy) ----------
    // B layout: lane ln col N=token; elem e<8 -> K(d) = dc*32 + half*8 + e ; e>=8 -> K(d) = dc*32 + 16 + half*8 + (e-8)
    v8f s[2];
    #pragma unroll
    for (int tg = 0; tg < 2; ++tg) {
      const int pg = tg ? pg1 : pg0;
      const float* krow = kp + (((size_t)kv * NPAGES + pg) * TPP + ln) * DIM + half * 8;
      v8f acc = vzero;
      #pragma unroll
      for (int dc = 0; dc < 4; ++dc) {
        f32x4 r0 = *(const f32x4*)(krow + dc * 32);
        f32x4 r1 = *(const f32x4*)(krow + dc * 32 + 4);
        f32x4 r2 = *(const f32x4*)(krow + dc * 32 + 16);
        f32x4 r3 = *(const f32x4*)(krow + dc * 32 + 20);
        v16bf bk;
        #pragma unroll
        for (int e = 0; e < 4; ++e) {
          bk[e]      = (__bf16)r0[e];
          bk[e + 4]  = (__bf16)r1[e];
          bk[e + 8]  = (__bf16)r2[e];
          bk[e + 12] = (__bf16)r3[e];
        }
        acc = __builtin_amdgcn_wmma_f32_16x16x32_bf16(false, aq[dc], false, bk,
                                                      (short)0, acc, false, false);
      }
      s[tg] = acc;
    }

    // ---------- mask + online softmax (C layout: col=token=ln, row=r+half*8) ----------
    const int tok0 = cbase + ln;
    const int tok1 = cbase + 16 + ln;
    #pragma unroll
    for (int r = 0; r < 8; ++r) {
      s[0][r] = (tok0 < kvlen) ? s[0][r] : NEG_BIG;
      s[1][r] = (tok1 < kvlen) ? s[1][r] : NEG_BIG;
    }

    float scale[8];
    #pragma unroll
    for (int r = 0; r < 8; ++r) {
      float x = fmaxf(s[0][r], s[1][r]);
      #pragma unroll
      for (int off = 8; off >= 1; off >>= 1)
        x = fmaxf(x, __shfl_xor(x, off, 32));
      const float mnew = fmaxf(m_run[r], x);
      scale[r] = __expf(m_run[r] - mnew);
      m_run[r] = mnew;
      const float p0 = __expf(s[0][r] - mnew);
      const float p1 = __expf(s[1][r] - mnew);
      float rs = p0 + p1;
      #pragma unroll
      for (int off = 8; off >= 1; off >>= 1)
        rs += __shfl_xor(rs, off, 32);
      l_run[r] = l_run[r] * scale[r] + rs;
      s[0][r] = p0;
      s[1][r] = p1;
    }

    // ---------- transpose P (C layout -> A layout) via per-wave LDS [row][tok] tile ----------
    #pragma unroll
    for (int r = 0; r < 8; ++r) {
      p_lds[wave][half * 8 + r][ln]      = (__bf16)s[0][r];
      p_lds[wave][half * 8 + r][16 + ln] = (__bf16)s[1][r];
    }
    __builtin_amdgcn_wave_barrier();   // DS in-order per wave; stop compiler reordering
    // A-frag row = ln, tokens: [half*8 .. half*8+7] and [16+half*8 .. 16+half*8+7] -> 2x b128
    const v8bf plo = *(const v8bf*)&p_lds[wave][ln][half * 8];
    const v8bf phi = *(const v8bf*)&p_lds[wave][ln][16 + half * 8];
    v16bf ap;
    #pragma unroll
    for (int e = 0; e < 8; ++e) { ap[e] = plo[e]; ap[e + 8] = phi[e]; }
    __builtin_amdgcn_wave_barrier();

    // ---------- rescale accumulators ----------
    #pragma unroll
    for (int nc = 0; nc < 8; ++nc)
      #pragma unroll
      for (int r = 0; r < 8; ++r) o_acc[nc][r] *= scale[r];

    // ---------- wait for V tile, then O += P * V ----------
    asm volatile("s_wait_asynccnt 0x0" ::: "memory");
    // V B-fragment: lane ln col N=d=nc*16+ln; elem e<8 -> token = half*8 + e ; e>=8 -> token = 16 + half*8 + (e-8)
    #pragma unroll
    for (int nc = 0; nc < 8; ++nc) {
      v16bf bv;
      #pragma unroll
      for (int e = 0; e < 16; ++e)
        bv[e] = (__bf16)v_stage[wave][(e < 8 ? half * 8 + e : 8 + half * 8 + e)][nc * 16 + ln];
      o_acc[nc] = __builtin_amdgcn_wmma_f32_16x16x32_bf16(false, ap, false, bv,
                                                          (short)0, o_acc[nc], false, false);
    }
  }

  // ---------- per-wave partials to LDS ----------
  if (lane == 0) {
    #pragma unroll
    for (int g = 0; g < GRP; ++g) {
      ml_lds[wave][0][g] = m_run[g];
      ml_lds[wave][1][g] = l_run[g];
    }
  }
  if (half == 0) {
    #pragma unroll
    for (int nc = 0; nc < 8; ++nc)
      #pragma unroll
      for (int g = 0; g < GRP; ++g)
        o_lds[wave][g][nc * 16 + ln] = o_acc[nc][g];
  }
  __syncthreads();

  // ---------- combine 8 wave partials (log-sum-exp merge) ----------
  for (int idx = threadIdx.x; idx < GRP * DIM; idx += 256) {
    const int g = idx / DIM;
    const int d = idx % DIM;
    float mt = -INFINITY;
    #pragma unroll
    for (int w = 0; w < NWAVES; ++w) mt = fmaxf(mt, ml_lds[w][0][g]);
    float lt = 0.0f, val = 0.0f;
    #pragma unroll
    for (int w = 0; w < NWAVES; ++w) {
      const float f = __expf(ml_lds[w][0][g] - mt);
      lt  += ml_lds[w][1][g] * f;
      val += o_lds[w][g][d] * f;
    }
    out[((size_t)b * NUM_H + kv * GRP + g) * DIM + d] = val / lt;
  }
}

extern "C" void kernel_launch(void* const* d_in, const int* in_sizes, int n_in,
                              void* d_out, int out_size, void* d_ws, size_t ws_size,
                              hipStream_t stream) {
  (void)in_sizes; (void)n_in; (void)out_size; (void)d_ws; (void)ws_size;
  const float* q   = (const float*)d_in[0];
  const float* kp  = (const float*)d_in[1];
  const float* vp  = (const float*)d_in[2];
  const int*   kvl = (const int*)d_in[3];
  const int*   pid = (const int*)d_in[4];
  float* out = (float*)d_out;
  dim3 grid(NUM_B * NKV);
  dim3 block(256);
  hipLaunchKernelGGL(paged_attn_wmma, grid, block, 0, stream, q, kp, vp, kvl, pid, out);
}
